// CausalNCMomentumAttention_22265110463166
// MI455X (gfx1250) — matrix operand order repaired and checked by reading, not compiled, not run on hardware
//
#include <hip/hip_runtime.h>
#include <hip/hip_bf16.h>
#include <math.h>

// ---------------------------------------------------------------------------
// Problem constants (from reference): N=2, L=2048, H=8, E=M=64
// ---------------------------------------------------------------------------
#define N_BATCH 2
#define SEQ_L   2048
#define HEADS   8
#define EDIM    64
#define MDIM    64
#define CHUNK   64
#define NCHUNK  (SEQ_L / CHUNK)
#define EPS     1e-6f
#define STEPSZ  1.0f
#define DELTA   0.1f

typedef __attribute__((ext_vector_type(16))) __bf16 v16bf;
typedef __attribute__((ext_vector_type(8)))  float  v8f;
typedef __attribute__((ext_vector_type(4)))  unsigned int v4u;
typedef __attribute__((ext_vector_type(8)))  int v8i_;
typedef __attribute__((ext_vector_type(4)))  int v4i_;

#define LDH 72   // LDS row stride in bf16 (64 + 8 pad -> every row 16B aligned)

#ifndef __has_builtin
#define __has_builtin(x) 0
#endif
#if __has_builtin(__builtin_amdgcn_tensor_load_to_lds)
#define HAVE_TDM 1
#else
#define HAVE_TDM 0
#endif

// ---------------------------------------------------------------------------
// WMMA operand loaders (wave32); contiguous k order (k = 16h + s), same in A and B.  All sources are staged
// so every loader reads CONTIGUOUS 16-half runs (lowers to ds_load_b128).
// A (16x32 bf16): lane l holds row (l&15); lanes 0-15 -> K 0..15,
//                 lanes 16-31 -> K 16..31.
// B (32x16 bf16): lane l holds col (l&15); lanes 0-15 K 0..15, 16-31 K 16..31.
// C/D (16x16 f32): VGPR r: lanes 0-15 -> row r, lanes 16-31 -> row r+8.
// ---------------------------------------------------------------------------
__device__ __forceinline__ v16bf load_A_rm(const __bf16* p, int row0, int k0,
                                           int lane) {
  int r  = row0 + (lane & 15);
  int kb = (lane < 16) ? 0 : 16;
  const __bf16* rp = p + r * LDH + k0 + kb;
  v16bf a;
#pragma unroll
  for (int j = 0; j < 8; ++j) a[j] = rp[j];         // K = kb .. kb+7
#pragma unroll
  for (int j = 0; j < 8; ++j) a[8 + j] = rp[8 + j]; // K = kb+8 .. kb+15
  return a;
}

// B operand where memory holds element(k,n) at p[n*LDH + k]  (contiguous)
__device__ __forceinline__ v16bf load_B_tr(const __bf16* p, int k0, int n0,
                                           int lane) {
  int n  = n0 + (lane & 15);
  int kb = (lane < 16) ? 0 : 16;
  const __bf16* rp = p + n * LDH + k0 + kb;
  v16bf b;
#pragma unroll
  for (int j = 0; j < 16; ++j) b[j] = rp[j];
  return b;
}

__device__ __forceinline__ v8f wmma_bf16(v16bf a, v16bf b, v8f c) {
  return __builtin_amdgcn_wmma_f32_16x16x32_bf16(false, a, false, b,
                                                 (short)0, c, false, false);
}

__device__ __forceinline__ float elu1(float x) {
  return (x > 0.0f) ? (x + 1.0f) : expf(x);   // elu(x)+1
}

// ---------------------------------------------------------------------------
// Tensor Data Mover: 1-D tile of 4096 bf16 (one 64x64 chunk, contiguous in
// global) -> LDS with 8-half row padding (pad every 32 DWORDs by 4 DWORDs,
// ISA 8.4: pad_interval code 4, pad_amount code 3) => LDS row stride 72.
// D# built per ISA 8.3/8.4; count=1, type=2 ("image"), data_size=1 (2 bytes).
// ---------------------------------------------------------------------------
#if HAVE_TDM
__device__ __forceinline__ void tdm_load_tile(const void* gsrc, void* ldst) {
  unsigned long long ga = (unsigned long long)(uintptr_t)gsrc;
  unsigned int la = (unsigned int)(uintptr_t)ldst;   // low 32 bits = LDS addr
  v4u g0;
  g0[0] = 1u;                                        // count=1, user desc
  g0[1] = la;                                        // lds_addr
  g0[2] = (unsigned int)(ga & 0xFFFFFFFFu);          // global_addr[31:0]
  g0[3] = (unsigned int)((ga >> 32) & 0x1FFFFFFu)    // global_addr[56:32]
          | (2u << 30);                               // type=2
  v8i_ g1;
  g1[0] = (1 << 16)      // data_size = 2 bytes
        | (1 << 20)      // pad_enable
        | (4 << 22)      // pad_interval: 32 DWORDs
        | (3 << 25);     // pad_amount:   4 DWORDs
  g1[1] = (int)(4096u << 16);  // tensor_dim0[15:0]=4096 (bits 63:48)
  g1[2] = 0;                   // tensor_dim0 hi = 0, tensor_dim1 lo = 0
  g1[3] = (int)(4096u << 16);  // tile_dim0 = 4096 (bits 127:112)
  g1[4] = 0;                   // tile_dim1 = tile_dim2 = 0 (unused)
  g1[5] = 4096;                // tensor_dim0_stride lo
  g1[6] = 0;
  g1[7] = 0;
  v4i_ z = {0, 0, 0, 0};
#if defined(__clang_major__) && (__clang_major__ >= 23)
  v8i_ z8 = {0, 0, 0, 0, 0, 0, 0, 0};
  __builtin_amdgcn_tensor_load_to_lds(g0, g1, z, z, z8, 0);
#else
  __builtin_amdgcn_tensor_load_to_lds(g0, g1, z, z, 0);
#endif
}

__device__ __forceinline__ void tdm_wait() {
#if __has_builtin(__builtin_amdgcn_s_wait_tensorcnt)
  __builtin_amdgcn_s_wait_tensorcnt(0);
#else
  asm volatile("s_wait_tensorcnt 0x0" ::: "memory");
#endif
}
#endif  // HAVE_TDM

// ---------------------------------------------------------------------------
// Kernel 1: feature map + bf16 repack (head-major) + rank-1 momentum scalars.
// One wave per (n, l, h).
// ---------------------------------------------------------------------------
__global__ __launch_bounds__(256) void prep_kernel(
    const float* __restrict__ q, const float* __restrict__ k,
    const float* __restrict__ v, const float* __restrict__ mask,
    __bf16* __restrict__ phiQ, __bf16* __restrict__ phiK,
    __bf16* __restrict__ Vb, float* __restrict__ kn2, float* __restrict__ vn2,
    float* __restrict__ kdot, float* __restrict__ vdot) {
  int wave = blockIdx.x * 8 + (threadIdx.x >> 5);
  int lane = threadIdx.x & 31;
  int n = wave / (SEQ_L * HEADS);
  int rem = wave % (SEQ_L * HEADS);
  int l = rem / HEADS;
  int h = rem % HEADS;

  size_t gidx = (((size_t)n * SEQ_L + l) * HEADS + h) * EDIM;   // [N,L,H,E]
  size_t widx = (((size_t)(n * HEADS + h)) * SEQ_L + l) * EDIM; // [N,H,L,E]
  float m  = mask[n * SEQ_L + l];
  float mp = (l > 0) ? mask[n * SEQ_L + l - 1] : 0.0f;
  size_t gprev = (l > 0) ? (gidx - (size_t)HEADS * EDIM) : gidx;

  float k2 = 0.f, v2 = 0.f, kd = 0.f, vd = 0.f;
#pragma unroll
  for (int j = 0; j < 2; ++j) {
    int e = lane + j * 32;
    float phiq = elu1(q[gidx + e]);
    phiQ[widx + e] = (__bf16)phiq;
    float phik = elu1(k[gidx + e]) * m;
    phiK[widx + e] = (__bf16)phik;
    float vx = v[gidx + e];
    Vb[widx + e] = (__bf16)vx;
    k2 += phik * phik;
    v2 += vx * vx;
    float phikp = elu1(k[gprev + e]) * mp;
    kd += phik * phikp;
    vd += vx * v[gprev + e];
  }
#pragma unroll
  for (int off = 16; off > 0; off >>= 1) {
    k2 += __shfl_down(k2, off, 32);
    v2 += __shfl_down(v2, off, 32);
    kd += __shfl_down(kd, off, 32);
    vd += __shfl_down(vd, off, 32);
  }
  if (lane == 0) {
    int o = (n * HEADS + h) * SEQ_L + l;
    kn2[o] = k2;
    vn2[o] = v2;
    kdot[o] = kd;
    vdot[o] = vd;
  }
}

// ---------------------------------------------------------------------------
// Kernel 2: momentum weights (backward recurrence) + Kn output.
// One block per (n,h).  ||k v^T||_F^2 = ||k||^2 ||v||^2 (rank-1 identity).
// ---------------------------------------------------------------------------
__global__ __launch_bounds__(256) void momentum_kernel(
    const float* __restrict__ keys, const float* __restrict__ mask,
    const float* __restrict__ kn2, const float* __restrict__ vn2,
    const float* __restrict__ kdot, const float* __restrict__ vdot,
    float* __restrict__ kn_out) {
  __shared__ float sk2[SEQ_L], sv2[SEQ_L], skd[SEQ_L], svd[SEQ_L], sa[SEQ_L];
  int b = blockIdx.x;
  int n = b / HEADS;
  int h = b % HEADS;
  int base = b * SEQ_L;
  for (int t = threadIdx.x; t < SEQ_L; t += 256) {
    sk2[t] = kn2[base + t];
    sv2[t] = vn2[base + t];
    skd[t] = kdot[base + t];
    svd[t] = vdot[base + t];
  }
  __syncthreads();
  if (threadIdx.x == 0) {
    float a = 1.0f;
    sa[SEQ_L - 1] = 1.0f;
    for (int t = SEQ_L - 2; t >= 0; --t) {
      float b2 = sk2[t] * sv2[t];
      float d2 = sk2[t + 1] * sv2[t + 1] + b2 - 2.0f * skd[t + 1] * svd[t + 1];
      d2 = fmaxf(d2, 0.0f);
      float bn = sqrtf(fmaxf(b2, 1e-30f));
      float dn = sqrtf(d2);
      float r  = sqrtf(dn / bn);
      float mu = (1.0f - r);
      mu = mu * mu;
      mu = fminf(fmaxf(mu, 0.0f), 1.0f - DELTA);
      a = 1.0f + mu * a;
      sa[t] = a;
    }
  }
  __syncthreads();
  for (int i = threadIdx.x; i < SEQ_L * EDIM; i += 256) {
    int t = i >> 6;
    int e = i & 63;
    size_t gi = (((size_t)n * SEQ_L + t) * HEADS + h) * EDIM + e;
    float phi = elu1(keys[gi]) * mask[n * SEQ_L + t];
    kn_out[gi] = phi * sa[t] * STEPSZ;
  }
}

// ---------------------------------------------------------------------------
// Kernel 3: chunked causal linear attention via WMMA.  One block per (n,h),
// 8 waves.  Q,K staged row-major by the TDM (with LDS row padding); K,V also
// staged transposed so every WMMA operand load is a contiguous b128 DS load.
// State S (E x M) kept f32 in LDS, mirrored transposed in bf16 (sSbt).
// ---------------------------------------------------------------------------
__global__ __launch_bounds__(256) void attn_kernel(
    const __bf16* __restrict__ phiQ, const __bf16* __restrict__ phiK,
    const __bf16* __restrict__ Vb, float* __restrict__ out) {
  __shared__ __align__(16) __bf16 sQ[CHUNK * LDH];   // [t][e]
  __shared__ __align__(16) __bf16 sK[CHUNK * LDH];   // [s][e]
  __shared__ __align__(16) __bf16 sKt[EDIM * LDH];   // [e][s]
  __shared__ __align__(16) __bf16 sVt[MDIM * LDH];   // [m][s]
  __shared__ __align__(16) __bf16 sP[CHUNK * LDH];   // [t][s]
  __shared__ __align__(16) __bf16 sSbt[MDIM * LDH];  // [m][e]  (S transposed)
  __shared__ float sS[EDIM * 65];
  __shared__ float svec[EDIM];
  __shared__ float zrow[CHUNK];

  int tid  = threadIdx.x;
  int wave = tid >> 5;
  int lane = tid & 31;
  int b = blockIdx.x;        // n*H + h
  int n = b / HEADS;
  int h = b % HEADS;
  size_t headbase = (size_t)b * SEQ_L * EDIM;

  for (int i = tid; i < EDIM * 65; i += 256) sS[i] = 0.0f;
  for (int i = tid; i < MDIM * LDH; i += 256) sSbt[i] = (__bf16)0.0f;
  if (tid < EDIM) svec[tid] = 0.0f;
  __syncthreads();

  for (int c = 0; c < NCHUNK; ++c) {
    const __bf16* gQ = phiQ + headbase + (size_t)c * CHUNK * EDIM;
    const __bf16* gK = phiK + headbase + (size_t)c * CHUNK * EDIM;
    const __bf16* gV = Vb   + headbase + (size_t)c * CHUNK * EDIM;

    // ---- stage row-major Q,K: Tensor Data Mover (pad -> stride 72) ----
#if HAVE_TDM
    if (wave == 0) {
      tdm_load_tile(gQ, sQ);
      tdm_wait();
    } else if (wave == 1) {
      tdm_load_tile(gK, sK);
      tdm_wait();
    }
#else
#pragma unroll
    for (int rep = 0; rep < 2; ++rep) {
      int i = tid * 8 + rep * 2048;
      int r = i >> 6, cc = i & 63;
      *(uint4*)&sQ[r * LDH + cc] = *(const uint4*)&gQ[i];
      *(uint4*)&sK[r * LDH + cc] = *(const uint4*)&gK[i];
    }
#endif
    // ---- stage transposed K^T and V^T (register-mediated scatter) ----
#pragma unroll
    for (int rep = 0; rep < 2; ++rep) {
      int i = tid * 8 + rep * 2048;
      int r = i >> 6, c0 = i & 63;
      uint4 kv = *(const uint4*)&gK[i];
      uint4 vv = *(const uint4*)&gV[i];
      const __bf16* kp = (const __bf16*)&kv;
      const __bf16* vp = (const __bf16*)&vv;
#pragma unroll
      for (int j = 0; j < 8; ++j) {
        sKt[(c0 + j) * LDH + r] = kp[j];
        sVt[(c0 + j) * LDH + r] = vp[j];
      }
    }
    if (c + 1 < NCHUNK) {                      // prefetch next chunk
      __builtin_prefetch(gQ + CHUNK * EDIM + tid * 16, 0, 1);
      __builtin_prefetch(gK + CHUNK * EDIM + tid * 16, 0, 1);
      __builtin_prefetch(gV + CHUNK * EDIM + tid * 16, 0, 1);
    }
    __syncthreads();

    // ---- P = Q Kc^T with causal mask, stored bf16 row-major ----
#pragma unroll
    for (int rep = 0; rep < 2; ++rep) {
      int tile = wave + rep * 8;               // 0..15 -> 4x4 tile grid
      int ti = tile >> 2, si = tile & 3;
      int r0 = ti * 16, s0 = si * 16;
      int col = s0 + (lane & 15);
      int ofs = (lane < 16) ? 0 : 8;
      if (si > ti) {
#pragma unroll
        for (int r = 0; r < 8; ++r)
          sP[(r0 + ofs + r) * LDH + col] = (__bf16)0.0f;
      } else {
        v8f acc = {};
#pragma unroll
        for (int k0 = 0; k0 < EDIM; k0 += 32) {
          v16bf a  = load_A_rm(sQ, r0, k0, lane);
          v16bf bb = load_B_tr(sK, k0, s0, lane);  // B(e,s) = K[s][e]
          acc = wmma_bf16(a, bb, acc);
        }
#pragma unroll
        for (int r = 0; r < 8; ++r) {
          int row = r0 + ofs + r;
          float pv = acc[r];
          if (col > row) pv = 0.0f;            // causal mask, diagonal tile
          sP[row * LDH + col] = (__bf16)pv;
        }
      }
    }
    __syncthreads();

    // ---- z_t = 1 / (rowsum(maskedP) + q_t . svec + eps) ----
    if (tid < CHUNK) {
      float rs = 0.0f, qs = 0.0f;
#pragma unroll 8
      for (int s = 0; s < CHUNK; ++s) rs += (float)sP[tid * LDH + s];
#pragma unroll 8
      for (int e = 0; e < EDIM; ++e) qs += (float)sQ[tid * LDH + e] * svec[e];
      zrow[tid] = 1.0f / (rs + qs + EPS);
    }
    __syncthreads();

    // ---- O = Q @ S + P @ V, scaled by z, direct to global ----
#pragma unroll
    for (int rep = 0; rep < 2; ++rep) {
      int tile = wave + rep * 8;
      int ti = tile >> 2, mj = tile & 3;
      int r0 = ti * 16, m0 = mj * 16;
      v8f acc = {};
#pragma unroll
      for (int k0 = 0; k0 < EDIM; k0 += 32) {
        v16bf a  = load_A_rm(sQ, r0, k0, lane);
        v16bf bb = load_B_tr(sSbt, k0, m0, lane);  // B(e,m) = S[e][m]
        acc = wmma_bf16(a, bb, acc);
      }
#pragma unroll
      for (int k0 = 0; k0 < CHUNK; k0 += 32) {
        v16bf a  = load_A_rm(sP, r0, k0, lane);
        v16bf bb = load_B_tr(sVt, k0, m0, lane);   // B(s,m) = V[s][m]
        acc = wmma_bf16(a, bb, acc);
      }
      int col = m0 + (lane & 15);
      int ofs = (lane < 16) ? 0 : 8;
#pragma unroll
      for (int r = 0; r < 8; ++r) {
        int row = r0 + ofs + r;
        int t = c * CHUNK + row;
        out[(((size_t)n * SEQ_L + t) * HEADS + h) * MDIM + col] =
            acc[r] * zrow[row];
      }
    }
    __syncthreads();

    // ---- S += Kc^T @ V (f32 in LDS, bf16 transposed mirror) ----
#pragma unroll
    for (int rep = 0; rep < 2; ++rep) {
      int tile = wave + rep * 8;
      int ei = tile >> 2, mj = tile & 3;
      int e0 = ei * 16, m0 = mj * 16;
      int col = m0 + (lane & 15);
      int ofs = (lane < 16) ? 0 : 8;
      v8f acc;
#pragma unroll
      for (int r = 0; r < 8; ++r) acc[r] = sS[(e0 + ofs + r) * 65 + col];
#pragma unroll
      for (int k0 = 0; k0 < CHUNK; k0 += 32) {
        v16bf a  = load_A_rm(sKt, e0, k0, lane);   // A(e,s) = K[s][e]
        v16bf bb = load_B_tr(sVt, k0, m0, lane);
        acc = wmma_bf16(a, bb, acc);
      }
#pragma unroll
      for (int r = 0; r < 8; ++r) {
        int row = e0 + ofs + r;
        sS[row * 65 + col] = acc[r];
        sSbt[col * LDH + row] = (__bf16)acc[r];    // transposed mirror
      }
    }
    // ---- svec += column sums of Kc (rows of sKt: contiguous) ----
    if (tid < EDIM) {
      float s = 0.0f;
#pragma unroll 8
      for (int srow = 0; srow < CHUNK; ++srow)
        s += (float)sKt[tid * LDH + srow];
      svec[tid] += s;
    }
    __syncthreads();
  }
}

// ---------------------------------------------------------------------------
// Host-side launcher
// ---------------------------------------------------------------------------
extern "C" void kernel_launch(void* const* d_in, const int* in_sizes, int n_in,
                              void* d_out, int out_size, void* d_ws,
                              size_t ws_size, hipStream_t stream) {
  const float* q    = (const float*)d_in[0];
  const float* k    = (const float*)d_in[1];
  const float* v    = (const float*)d_in[2];
  const float* mask = (const float*)d_in[3];

  char* ws = (char*)d_ws;
  const size_t TENSOR = (size_t)N_BATCH * HEADS * SEQ_L * EDIM; // 2,097,152
  __bf16* phiQ = (__bf16*)ws;                       // 4 MiB
  __bf16* phiK = (__bf16*)(ws + TENSOR * 2);        // 4 MiB
  __bf16* Vb   = (__bf16*)(ws + TENSOR * 4);        // 4 MiB
  float* kn2  = (float*)(ws + TENSOR * 6);
  float* vn2  = kn2 + (size_t)N_BATCH * HEADS * SEQ_L;
  float* kdot = vn2 + (size_t)N_BATCH * HEADS * SEQ_L;
  float* vdot = kdot + (size_t)N_BATCH * HEADS * SEQ_L;

  float* out    = (float*)d_out;                    // [N,L,H,M]
  float* kn_out = out + TENSOR;                     // [N,L,H,E]

  dim3 g1((N_BATCH * SEQ_L * HEADS) / 8);
  prep_kernel<<<g1, 256, 0, stream>>>(q, k, v, mask, phiQ, phiK, Vb, kn2, vn2,
                                      kdot, vdot);

  momentum_kernel<<<dim3(N_BATCH * HEADS), 256, 0, stream>>>(
      k, mask, kn2, vn2, kdot, vdot, kn_out);

  attn_kernel<<<dim3(N_BATCH * HEADS), 256, 0, stream>>>(phiQ, phiK, Vb, out);
}
